// HumanPoseDiscriminator_11708080849262
// MI455X (gfx1250) — compile-verified, matrix operand order checked by reading, not run on hardware
//
#include <hip/hip_runtime.h>
#include <hip/hip_bf16.h>
#include <math.h>

// ---------------------------------------------------------------------------
// Fused pose-angle + 4-layer MLP discriminator for gfx1250 (MI455X).
//  - TDM (tensor_load_to_lds) double-buffered streaming of the 107MB input
//  - GEMMs on v_wmma_f32_16x16x32_f16, f32 accumulate
//  - Activations in LDS f16; weights transposed in LDS; act3/4 alias staging
// ---------------------------------------------------------------------------

typedef _Float16 v16h __attribute__((ext_vector_type(16)));
typedef _Float16 v8h  __attribute__((ext_vector_type(8)));
typedef float    v8f  __attribute__((ext_vector_type(8)));
typedef unsigned int uint32x4 __attribute__((ext_vector_type(4)));
typedef int      int32x4 __attribute__((ext_vector_type(4)));
typedef int      int32x8 __attribute__((ext_vector_type(8)));

#define RAD2DEG 57.29577951308232f

#if defined(__has_builtin)
#if __has_builtin(__builtin_amdgcn_tensor_load_to_lds) && \
    __has_builtin(__builtin_amdgcn_s_wait_tensorcnt)
#define HAVE_TDM 1
#endif
#endif
#ifndef HAVE_TDM
#define HAVE_TDM 0
#endif

union Frag16 { v16h v; v8h h8[2]; };

// ---------------------------------------------------------------------------
// TDM: async DMA of a [rows x 51] f32 tile (row stride 51 f32) into LDS.
// D# bit layout per CDNA5 ISA 8.3/8.4: group0 = {count,lds_addr,global_addr,
// type=2}; group1 = {data_size=4B, tensor_dim0=51, tensor_dim1=rows,
// tile_dim0=51, tile_dim1=rows, dim0_stride=51}. Groups 2/3 zero (2-D tile).
// ---------------------------------------------------------------------------
#if HAVE_TDM
__device__ __forceinline__ void tdm_load_tile(const float* gsrc, void* ldsDst,
                                              int rows) {
  const unsigned long long ga = (unsigned long long)(uintptr_t)gsrc;
  const unsigned int       la = (unsigned int)(uintptr_t)ldsDst;
  uint32x4 g0;
  g0.x = 1u;                                            // count=1 (user D#)
  g0.y = la;                                            // lds_addr
  g0.z = (unsigned int)(ga & 0xFFFFFFFFu);              // global_addr[31:0]
  g0.w = (unsigned int)((ga >> 32) & 0x1FFFFFFu) | (2u << 30);  // [56:32]|type=2
  int32x8 g1;
  g1[0] = (2 << 16);        // data_size = 2 -> 4 bytes
  g1[1] = (51 << 16);       // tensor_dim0[15:0] = 51
  g1[2] = (rows << 16);     // tensor_dim0[31:16]=0 | tensor_dim1[15:0]=rows
  g1[3] = (51 << 16);       // tensor_dim1[31:16]=0 | tile_dim0 = 51
  g1[4] = rows;             // tile_dim1 = rows, tile_dim2 = 0
  g1[5] = 51;               // tensor_dim0_stride[31:0] = 51
  g1[6] = (51 << 16);       // dim0_stride[47:32]=0 | tensor_dim1_stride lo = 51
  g1[7] = 0;
  const int32x4 z4 = {0, 0, 0, 0};
#if defined(__clang_major__) && (__clang_major__ >= 23)
  const int32x8 z8 = {0, 0, 0, 0, 0, 0, 0, 0};
  __builtin_amdgcn_tensor_load_to_lds(g0, g1, z4, z4, z8, 0);
#else
  __builtin_amdgcn_tensor_load_to_lds(g0, g1, z4, z4, 0);
#endif
}
#endif

// transformed pose: (x, y, z)_raw -> (x, z, -y)
__device__ __forceinline__ void getp(const float* __restrict__ r, int j,
                                     float& x, float& y, float& z) {
  x =  r[3 * j + 0];
  y =  r[3 * j + 2];
  z = -r[3 * j + 1];
}

__device__ __forceinline__ float ang_between_v(float ax, float ay, float az,
                                               float bx, float by, float bz) {
  float dot = ax * bx + ay * by + az * bz;
  float nn  = sqrtf((ax * ax + ay * ay + az * az) *
                    (bx * bx + by * by + bz * bz));
  float c = dot / nn;
  c = fminf(1.f, fmaxf(-1.f, c));
  return acosf(c) * RAD2DEG;
}

__device__ __forceinline__ float ang_z(const float* __restrict__ r, int j1, int j2) {
  float x1, y1, z1, x2, y2, z2;
  getp(r, j1, x1, y1, z1);
  getp(r, j2, x2, y2, z2);
  float ax = x1 - x2, ay = y1 - y2, az = z1 - z2;
  float c = az / sqrtf(ax * ax + ay * ay + az * az);
  c = fminf(1.f, fmaxf(-1.f, c));
  return acosf(c) * RAD2DEG;
}

__device__ __forceinline__ float ang3(const float* __restrict__ r, int j1, int j2, int j3) {
  float x1, y1, z1, x2, y2, z2, x3, y3, z3;
  getp(r, j1, x1, y1, z1);
  getp(r, j2, x2, y2, z2);
  getp(r, j3, x3, y3, z3);
  return ang_between_v(x1 - x2, y1 - y2, z1 - z2,
                       x3 - x2, y3 - y2, z3 - z2);
}

__device__ __forceinline__ float twistf(const float* __restrict__ r,
                                        int j1, int j2, int j3, int j4) {
  float x1, y1, z1, x2, y2, z2, x3, y3, z3, x4, y4, z4;
  getp(r, j1, x1, y1, z1);
  getp(r, j2, x2, y2, z2);
  getp(r, j3, x3, y3, z3);
  getp(r, j4, x4, y4, z4);
  return ang_between_v(x1 - x2, y1 - y2, z1 - z2,
                       x4 - x3, y4 - y3, z4 - z3);
}

// One GEMM layer for this wave's 16 rows (wave-private, no barriers needed):
//   actIn  [128 rows][K] f16, Wt [NOUT][K] f16, actOut [128 rows][NOUT] f16
// D = leaky(A*W + b); A/B frags each load as 2x contiguous v8h (ds_load_b128).
template <int K, int NOUT>
__device__ __forceinline__ void gemm_layer(const _Float16* __restrict__ actIn,
                                           const _Float16* __restrict__ Wt,
                                           const float* __restrict__ bias,
                                           _Float16* __restrict__ actOut,
                                           int rowOff, int lane) {
  const int l16  = lane & 15;
  const int half = lane >> 4;
  constexpr int KT = K / 32;

  Frag16 a[KT];
#pragma unroll
  for (int kt = 0; kt < KT; ++kt) {
    const _Float16* ab = actIn + (rowOff + l16) * K + kt * 32 + half * 8;
    a[kt].h8[0] = *(const v8h*)ab;
    a[kt].h8[1] = *(const v8h*)(ab + 16);
  }

#pragma unroll
  for (int nt = 0; nt < NOUT / 16; ++nt) {
    const int n = nt * 16 + l16;
    const float bv = bias[n];
    v8f acc = {bv, bv, bv, bv, bv, bv, bv, bv};
#pragma unroll
    for (int kt = 0; kt < KT; ++kt) {
      Frag16 b;
      const _Float16* wb = Wt + n * K + kt * 32 + half * 8;
      b.h8[0] = *(const v8h*)wb;
      b.h8[1] = *(const v8h*)(wb + 16);
      acc = __builtin_amdgcn_wmma_f32_16x16x32_f16(
          false, a[kt].v, false, b.v, (short)0, acc, false, false);
    }
#pragma unroll
    for (int i = 0; i < 8; ++i) {
      float x = acc[i];
      x = fmaxf(x, 0.2f * x);              // leaky ReLU in 2 VALU ops
      const int m = rowOff + i + half * 8; // C/D layout: VGPR i -> M=i+8*half
      actOut[m * NOUT + n] = (_Float16)x;
    }
  }
}

constexpr int BLOCK = 256;   // 8 wave32s
constexpr int ROWS  = 128;   // rows per tile (16 per wave)
constexpr int ITERS = 4;     // tiles per block -> 512 rows/block
constexpr int K1 = 96;       // 65 padded to 96
constexpr int N1 = 128;
constexpr int K2 = 128;
constexpr int N2 = 64;
constexpr int K3 = 64;
constexpr int N3 = 32;

__global__ __launch_bounds__(BLOCK, 2)
void pose_disc_kernel(const float* __restrict__ frames,
                      const float* __restrict__ W1, const float* __restrict__ b1,
                      const float* __restrict__ W2, const float* __restrict__ b2,
                      const float* __restrict__ W3, const float* __restrict__ b3,
                      const float* __restrict__ W4, const float* __restrict__ b4,
                      float* __restrict__ out) {
  // ---------------- LDS (~155 KB -> 2 workgroups / WGP of 320 KB) ----------
  __shared__ __align__(16) _Float16 sW1t[N1 * K1];      // 24576 B
  __shared__ __align__(16) _Float16 sW2t[N2 * K2];      // 16384 B
  __shared__ __align__(16) _Float16 sW3t[N3 * K3];      //  4096 B
  __shared__ __align__(16) _Float16 sW4[32];
  __shared__ float sB1[N1], sB2[N2], sB3[N3], sB4v[1];
  __shared__ __align__(16) _Float16 sAct1[ROWS * K1];   // 24576 B
  __shared__ __align__(16) _Float16 sAct2[ROWS * K2];   // 32768 B
  __shared__ __align__(16) float sStage[2][ROWS * 51];  // 2 x 26112 B (DMA dbl-buf)
  // sAct3/sAct4 alias the *current* staging buffer (dead after angle phase,
  // protected by the barrier before the GEMM phase; wave-private thereafter).

  const int tid  = threadIdx.x;
  const int wave = tid >> 5;
  const int lane = tid & 31;
  const int rowOff = wave * 16;
  const long blkRow0 = (long)blockIdx.x * (ROWS * ITERS);

#if HAVE_TDM
  if (wave == 0)  // kick off tile 0 DMA; overlaps with weight loading below
    tdm_load_tile(frames + blkRow0 * 51, &sStage[0][0], ROWS);
#endif

  // ---- load + transpose weights to f16 (tiny, L2-resident broadcast) ----
  for (int i = tid; i < N1 * K1; i += BLOCK) {
    const int n = i / K1, k = i - n * K1;
    sW1t[i] = (k < 65) ? (_Float16)W1[k * N1 + n] : (_Float16)0.f;
  }
  for (int i = tid; i < N2 * K2; i += BLOCK) {
    const int n = i >> 7, k = i & 127;
    sW2t[i] = (_Float16)W2[k * N2 + n];
  }
  for (int i = tid; i < N3 * K3; i += BLOCK) {
    const int n = i >> 6, k = i & 63;
    sW3t[i] = (_Float16)W3[k * N3 + n];
  }
  if (tid < 32)  sW4[tid] = (_Float16)W4[tid];
  if (tid < N1)  sB1[tid] = b1[tid];
  if (tid < N2)  sB2[tid] = b2[tid];
  if (tid < N3)  sB3[tid] = b3[tid];
  if (tid == 0)  sB4v[0]  = b4[0];
  __syncthreads();

  int cur = 0;
  for (int it = 0; it < ITERS; ++it) {
    const long rowBase = blkRow0 + (long)it * ROWS;

#if HAVE_TDM
    if (wave == 0) {
      // TDM ops of one wave complete in order; wait argument must be an
      // immediate, so split the two call sites.
      if (it + 1 < ITERS) {
        // prefetch next tile into the other buffer, then wait for tile `it`
        tdm_load_tile(frames + (rowBase + ROWS) * 51, &sStage[cur ^ 1][0], ROWS);
        __builtin_amdgcn_s_wait_tensorcnt(1);
      } else {
        __builtin_amdgcn_s_wait_tensorcnt(0);
      }
    }
#else
    {  // fallback: coalesced float4 block copy (region is 16B aligned)
      const float4* gsrc = (const float4*)(frames + rowBase * 51);
      float4* dst4 = (float4*)&sStage[cur][0];
      for (int i = tid; i < ROWS * 51 / 4; i += BLOCK) dst4[i] = gsrc[i];
      if (it + 1 < ITERS)
        __builtin_prefetch(frames + (rowBase + ROWS) * 51, 0, 0);
    }
#endif
    __syncthreads();  // staging[cur] complete & visible to all waves

    // --- per-row angle features + enriched f16 row into sAct1 [row][96] ---
    if (tid < ROWS) {
      const float* r = &sStage[cur][tid * 51];
      float ang[14];
      ang[0]  = ang_z(r, 11, 12) - 10.0f;        // shoulder_left_z
      ang[1]  = ang3(r, 12, 11, 14);             // shoulder_left_shoulderline
      ang[2]  = ang_z(r, 14, 15) - 10.0f;        // shoulder_right_z
      ang[3]  = ang3(r, 15, 14, 11);             // shoulder_right_shoulderline
      ang[4]  = 180.0f - ang3(r, 13, 12, 11);    // elbow_left
      ang[5]  = 180.0f - ang3(r, 14, 15, 16);    // elbow_right
      ang[6]  = 180.0f - ang3(r, 4, 5, 6);       // knee_left
      ang[7]  = 180.0f - ang3(r, 1, 2, 3);       // knee_right
      ang[8]  = 180.0f - ang_z(r, 0, 8);         // stoop
      ang[9]  = twistf(r, 1, 4, 11, 14);         // trunk_twist
      ang[10] = twistf(r, 1, 4, 0, 7);           // trunk_sidebending
      ang[11] = 180.0f - ang_z(r, 8, 10);        // neck
      ang[12] = twistf(r, 11, 14, 8, 10);        // neck_sidebending
      ang[13] = twistf(r, 11, 14, 9, 10);        // neck_twist

      _Float16* dst = &sAct1[tid * K1];
#pragma unroll
      for (int k = 0; k < 51; ++k) dst[k] = (_Float16)r[k];
#pragma unroll
      for (int k = 0; k < 14; ++k) dst[51 + k] = (_Float16)ang[k];
#pragma unroll
      for (int k = 65; k < K1; ++k) dst[k] = (_Float16)0.f;
    }
    __syncthreads();  // sAct1 ready; sStage[cur] now dead -> reuse as act3/4

    _Float16* sAct3 = (_Float16*)&sStage[cur][0];                  // 16384 B
    _Float16* sAct4 = (_Float16*)((char*)&sStage[cur][0] + 16384); //  8192 B

    // --- fused MLP: each wave owns 16 rows end-to-end (no inter-wave deps) ---
    gemm_layer<K1, N1>(sAct1, sW1t, sB1, sAct2, rowOff, lane);  // 65->128
    gemm_layer<K2, N2>(sAct2, sW2t, sB2, sAct3, rowOff, lane);  // 128->64
    gemm_layer<K3, N3>(sAct3, sW3t, sB3, sAct4, rowOff, lane);  // 64->32

    // --- final 32->1 dot + sigmoid: lanes 0..15 each finish one row ---
    if (lane < 16) {
      const int row = rowOff + lane;
      const _Float16* arow = &sAct4[row * 32];
      float s = sB4v[0];
#pragma unroll
      for (int k = 0; k < 32; ++k) s += (float)arow[k] * (float)sW4[k];
      out[rowBase + row] = 1.0f / (1.0f + __expf(-s));
    }
    __syncthreads();  // buffers recycled next tile
    cur ^= 1;
  }
}

extern "C" void kernel_launch(void* const* d_in, const int* in_sizes, int n_in,
                              void* d_out, int out_size, void* d_ws, size_t ws_size,
                              hipStream_t stream) {
  (void)n_in; (void)d_ws; (void)ws_size; (void)out_size;
  const float* frames = (const float*)d_in[0];
  const float* W1 = (const float*)d_in[1];
  const float* b1 = (const float*)d_in[2];
  const float* W2 = (const float*)d_in[3];
  const float* b2 = (const float*)d_in[4];
  const float* W3 = (const float*)d_in[5];
  const float* b3 = (const float*)d_in[6];
  const float* W4 = (const float*)d_in[7];
  const float* b4 = (const float*)d_in[8];
  float* out = (float*)d_out;

  const int Btot = in_sizes[0] / 51;        // 524288
  const int blocks = Btot / (ROWS * ITERS); // 1024
  pose_disc_kernel<<<blocks, BLOCK, 0, stream>>>(frames, W1, b1, W2, b2,
                                                 W3, b3, W4, b4, out);
}
